// GuidedCorrespondenceLoss_26474178412827
// MI455X (gfx1250) — compile-verified, matrix operand order checked by reading, not run on hardware
//
#include <hip/hip_runtime.h>
#include <math.h>

// ---------------- problem constants ----------------
#define EPSF      2.220446049250313e-16f
#define PATCH     7
#define STRIDE    3
#define FH        256           // feature H
#define FW        256           // feature W
#define HO        84            // unfold output H  ((256-7)/3+1)
#define WO        84            // unfold output W
#define CK        6272          // 128*49 channels after unfold
#define NP        4096          // 64*64 sample points
#define KM        64            // mask channels padded (49 -> 64)
#define INV49     (1.0f/49.0f)
#define PROGW     10.0f

#define BM        128           // GEMM block tile M
#define BN        64            // GEMM block tile N
#define BK        32            // GEMM K step
#define LDA       40            // padded LDS row stride (elements): 80B -> conflict-free

typedef __attribute__((ext_vector_type(16))) __bf16 v16bf;
typedef __attribute__((ext_vector_type(8)))  __bf16 v8bf;
typedef __attribute__((ext_vector_type(8)))  float  v8f;
typedef __attribute__((ext_vector_type(4)))  int    v4i;

typedef __attribute__((address_space(1))) v4i* gptr_v4i;   // global int4*
typedef __attribute__((address_space(3))) v4i* lptr_v4i;   // LDS    int4*

#if defined(__has_builtin)
#  if __has_builtin(__builtin_amdgcn_global_load_async_to_lds_b128)
#    define HAVE_ASYNC_B128 1
#  endif
#  if __has_builtin(__builtin_amdgcn_s_wait_asynccnt)
#    define HAVE_WAIT_ASYNC 1
#  endif
#endif

// 16-byte async copy global -> LDS (ASYNCcnt-tracked on gfx1250)
__device__ __forceinline__ void async_copy16(__bf16* ldst, const __bf16* gsrc) {
#if defined(HAVE_ASYNC_B128)
    __builtin_amdgcn_global_load_async_to_lds_b128(
        (gptr_v4i)gsrc, (lptr_v4i)ldst, 0, 0);
#else
    unsigned laddr = (unsigned)(uintptr_t)ldst;     // low 32 bits = LDS byte address
    asm volatile("global_load_async_to_lds_b128 %0, %1, off"
                 :: "v"(laddr), "v"((const void*)gsrc)
                 : "memory");
#endif
}

__device__ __forceinline__ void async_wait0() {
#if defined(HAVE_WAIT_ASYNC)
    __builtin_amdgcn_s_wait_asynccnt(0);
#else
    asm volatile("s_wait_asynccnt 0x0" ::: "memory");
#endif
}

// ---------------------------------------------------------------------------
// Kernel 0: zero the scalar output (harness poisons d_out with 0xAA)
// ---------------------------------------------------------------------------
__global__ void k_zero(float* out) { if (threadIdx.x == 0) out[0] = 0.0f; }

// ---------------------------------------------------------------------------
// Kernel 1: field -> patch base offsets (nearest grid sample into 84x84 grid)
// ---------------------------------------------------------------------------
__global__ void k_field_idx(const float* __restrict tfield,
                            const float* __restrict rfield,
                            int* __restrict tIdx, int* __restrict rIdx) {
    int p = blockIdx.x * blockDim.x + threadIdx.x;
    if (p >= NP) return;
    {
        float gx = tfield[p * 2 + 0], gy = tfield[p * 2 + 1];
        int ix = (int)fminf(fmaxf(roundf((gx + 1.0f) * 0.5f * (float)(WO - 1)), 0.0f), (float)(WO - 1));
        int iy = (int)fminf(fmaxf(roundf((gy + 1.0f) * 0.5f * (float)(HO - 1)), 0.0f), (float)(HO - 1));
        tIdx[p] = (iy * STRIDE) * FW + (ix * STRIDE);
    }
    {
        float gx = rfield[p * 2 + 0], gy = rfield[p * 2 + 1];
        int ix = (int)fminf(fmaxf(roundf((gx + 1.0f) * 0.5f * (float)(WO - 1)), 0.0f), (float)(WO - 1));
        int iy = (int)fminf(fmaxf(roundf((gy + 1.0f) * 0.5f * (float)(HO - 1)), 0.0f), (float)(HO - 1));
        rIdx[p] = (iy * STRIDE) * FW + (ix * STRIDE);
    }
}

// ---------------------------------------------------------------------------
// Kernel 2: refer per-channel mean over the 4096 gathered points
// ---------------------------------------------------------------------------
__global__ void k_refer_mean(const float* __restrict refer,
                             const int* __restrict rIdx,
                             float* __restrict ymean) {
    int cp = blockIdx.x * blockDim.x + threadIdx.x;
    if (cp >= CK) return;
    int c  = cp / 49;
    int r  = cp % 49;
    int off = c * (FH * FW) + (r / PATCH) * FW + (r % PATCH);
    float s = 0.0f;
    for (int p = 0; p < NP; ++p) s += refer[off + rIdx[p]];
    ymean[cp] = s * (1.0f / (float)NP);
}

// ---------------------------------------------------------------------------
// Kernel 3: gather a point's 6272-vector, subtract ymean, L2-normalize, bf16.
// ---------------------------------------------------------------------------
__global__ __launch_bounds__(256) void k_gather_norm(const float* __restrict feat,
                                                     const int* __restrict idx,
                                                     const float* __restrict ymean,
                                                     __bf16* __restrict outRow) {
    __shared__ float srow[CK];
    __shared__ float sred[256];
    const int p    = blockIdx.x;
    const int t    = threadIdx.x;
    const int base = idx[p];
    float ss = 0.0f;
    for (int cp = t; cp < CK; cp += 256) {
        int c = cp / 49, r = cp % 49;
        float v = feat[c * (FH * FW) + base + (r / PATCH) * FW + (r % PATCH)];
        float cv = v - ymean[cp];
        srow[cp] = cv;
        ss += cv * cv;
    }
    sred[t] = ss;
    __syncthreads();
    for (int s = 128; s > 0; s >>= 1) {
        if (t < s) sred[t] += sred[t + s];
        __syncthreads();
    }
    const float inv = 1.0f / (sqrtf(sred[0]) + EPSF);
    __bf16* row = outRow + (size_t)p * CK;
    for (int cp = t; cp < CK; cp += 256) row[cp] = (__bf16)(srow[cp] * inv);
}

// ---------------------------------------------------------------------------
// Kernel 4: mask patch gather (49 channels, padded to 64) + squared sums.
// ---------------------------------------------------------------------------
__global__ void k_mask_patches(const float* __restrict mask,
                               const int* __restrict tIdx, const int* __restrict rIdx,
                               __bf16* __restrict Tm, __bf16* __restrict Rm,
                               float* __restrict xs, float* __restrict ys) {
    int p = blockIdx.x * blockDim.x + threadIdx.x;
    if (p >= NP) return;
    {
        int base = tIdx[p];
        float s = 0.0f;
        __bf16* row = Tm + (size_t)p * KM;
        for (int r = 0; r < 49; ++r) {
            float v = mask[base + (r / PATCH) * FW + (r % PATCH)];
            s += v * v;
            row[r] = (__bf16)v;
        }
        for (int r = 49; r < KM; ++r) row[r] = (__bf16)0.0f;
        xs[p] = s;
    }
    {
        int base = rIdx[p];
        float s = 0.0f;
        __bf16* row = Rm + (size_t)p * KM;
        for (int r = 0; r < 49; ++r) {
            float v = mask[base + (r / PATCH) * FW + (r % PATCH)];
            s += v * v;
            row[r] = (__bf16)v;
        }
        for (int r = 49; r < KM; ++r) row[r] = (__bf16)0.0f;
        ys[p] = s;
    }
}

// ---------------------------------------------------------------------------
// Fragment loaders (ISA 7.12.2 16-bit A 16x32 layout: two 16B chunks/lane).
// ---------------------------------------------------------------------------
__device__ __forceinline__ v16bf pack_frag(v8bf lo, v8bf hi) {
    v16bf r;
#pragma unroll
    for (int i = 0; i < 8; ++i) { r[i] = lo[i]; r[i + 8] = hi[i]; }
    return r;
}

// from global, row-major [rows x ldk]
__device__ __forceinline__ v16bf gbl_frag(const __bf16* __restrict m,
                                          int rowBase, int kb, int ldk, int lane) {
    const __bf16* p = m + (size_t)(rowBase + (lane & 15)) * ldk + kb + ((lane & 16) ? 8 : 0);
    return pack_frag(*(const v8bf*)(p), *(const v8bf*)(p + 16));
}

// from LDS tile, padded row stride LDA
__device__ __forceinline__ v16bf lds_frag(const __bf16* __restrict tile,
                                          int rowBase, int lane) {
    const __bf16* p = tile + (rowBase + (lane & 15)) * LDA + ((lane & 16) ? 8 : 0);
    return pack_frag(*(const v8bf*)(p), *(const v8bf*)(p + 16));
}

#define WMMA_BF16(A, B, C) \
    __builtin_amdgcn_wmma_f32_16x16x32_bf16(false, (A), false, (B), (short)0, (C), false, false)

// ---------------------------------------------------------------------------
// Kernel 5: fused distance GEMM with async double-buffered LDS staging.
// Block tile 128(M) x 64(N), 8 waves in a 4x2 grid, wave tile 32x32.
//   accS: cosine similarity over K=6272 (LDS-staged bf16, async pipeline)
//   accA: mask-patch dot over K=64 (direct global fragments)
// Epilogue: d = relu((1-sim)/2) + 10*relu((xs - 2A + ys)/49) -> Dm[4096x4096]
// ---------------------------------------------------------------------------
__global__ __launch_bounds__(256) void k_dist_gemm(const __bf16* __restrict Xn,
                                                   const __bf16* __restrict Yn,
                                                   const __bf16* __restrict Tm,
                                                   const __bf16* __restrict Rm,
                                                   const float* __restrict xs,
                                                   const float* __restrict ys,
                                                   float* __restrict Dm) {
    __shared__ __bf16 As[2][BM * LDA];   // 2 x 10240 B
    __shared__ __bf16 Bs[2][BN * LDA];   // 2 x  5120 B

    const int t    = threadIdx.x;
    const int lane = t & 31;
    const int wave = t >> 5;
    const int wm   = wave >> 1;                  // 0..3  -> M strip
    const int wn   = wave & 1;                   // 0..1  -> N strip
    const int tM   = blockIdx.y * BM + wm * 32;  // global M origin of wave
    const int tN   = blockIdx.x * BN + wn * 32;  // global N origin of wave

    const __bf16* Ablk = Xn + (size_t)blockIdx.y * BM * CK;
    const __bf16* Bblk = Yn + (size_t)blockIdx.x * BN * CK;

    // async stage of one K-slab into LDS buffer `buf`
    auto issue_slab = [&](int buf, int kb) {
        {   // A: 128 rows x 4 chunks = 512 chunks, 2 per thread
            int id = t;
            int row = id >> 2, c = (id & 3) * 8;
            async_copy16(&As[buf][row * LDA + c], Ablk + (size_t)row * CK + kb + c);
            id = t + 256;
            row = id >> 2; c = (id & 3) * 8;
            async_copy16(&As[buf][row * LDA + c], Ablk + (size_t)row * CK + kb + c);
        }
        {   // B: 64 rows x 4 chunks = 256 chunks, 1 per thread
            int row = t >> 2, c = (t & 3) * 8;
            async_copy16(&Bs[buf][row * LDA + c], Bblk + (size_t)row * CK + kb + c);
        }
    };

    v8f accS[2][2], accA[2][2];
#pragma unroll
    for (int mi = 0; mi < 2; ++mi)
#pragma unroll
        for (int ni = 0; ni < 2; ++ni)
#pragma unroll
            for (int e = 0; e < 8; ++e) { accS[mi][ni][e] = 0.0f; accA[mi][ni][e] = 0.0f; }

    // ---- small mask GEMM: K = 64 (two 32-steps), direct global fragments ----
#pragma unroll
    for (int kb = 0; kb < KM; kb += 32) {
        v16bf a0 = gbl_frag(Tm, tM,      kb, KM, lane);
        v16bf a1 = gbl_frag(Tm, tM + 16, kb, KM, lane);
        v16bf b0 = gbl_frag(Rm, tN,      kb, KM, lane);
        v16bf b1 = gbl_frag(Rm, tN + 16, kb, KM, lane);
        accA[0][0] = WMMA_BF16(a0, b0, accA[0][0]);
        accA[0][1] = WMMA_BF16(a0, b1, accA[0][1]);
        accA[1][0] = WMMA_BF16(a1, b0, accA[1][0]);
        accA[1][1] = WMMA_BF16(a1, b1, accA[1][1]);
    }

    // ---- main cosine GEMM: K = 6272, async double-buffered pipeline ----
    issue_slab(0, 0);
    int buf = 0;
    for (int kb = 0; kb < CK; kb += BK) {
        async_wait0();          // this wave's slab landed in LDS
        __syncthreads();        // every wave's slab landed; prev buffer free
        if (kb + BK < CK) issue_slab(buf ^ 1, kb + BK);   // overlap next fill

        v16bf a0 = lds_frag(As[buf], wm * 32,      lane);
        v16bf a1 = lds_frag(As[buf], wm * 32 + 16, lane);
        v16bf b0 = lds_frag(Bs[buf], wn * 32,      lane);
        v16bf b1 = lds_frag(Bs[buf], wn * 32 + 16, lane);
        accS[0][0] = WMMA_BF16(a0, b0, accS[0][0]);
        accS[0][1] = WMMA_BF16(a0, b1, accS[0][1]);
        accS[1][0] = WMMA_BF16(a1, b0, accS[1][0]);
        accS[1][1] = WMMA_BF16(a1, b1, accS[1][1]);
        buf ^= 1;
    }

    // ---- epilogue: C/D layout (VGPR v -> M = v + 8*(lane>=16), N = lane&15) ----
#pragma unroll
    for (int mi = 0; mi < 2; ++mi) {
#pragma unroll
        for (int ni = 0; ni < 2; ++ni) {
            const int gm0 = tM + mi * 16 + ((lane >> 4) << 3);
            const int gn  = tN + ni * 16 + (lane & 15);
            const float ysn = ys[gn];
#pragma unroll
            for (int v = 0; v < 8; ++v) {
                const int gm = gm0 + v;
                const float sim = accS[mi][ni][v];
                const float l2  = (xs[gm] - 2.0f * accA[mi][ni][v] + ysn) * INV49;
                const float d   = fmaxf(0.0f, (1.0f - sim) * 0.5f) + PROGW * fmaxf(0.0f, l2);
                Dm[(size_t)gm * NP + gn] = d;
            }
        }
    }
}

// ---------------------------------------------------------------------------
// Kernel 6: per-row reduction:  dmin -> w = exp((1 - d/(dmin+eps))/0.5)
// CX = max(w)/sum(w);  loss += -log(CX)/NP  (atomic)
// ---------------------------------------------------------------------------
__global__ __launch_bounds__(256) void k_row_reduce(const float* __restrict Dm,
                                                    float* __restrict loss) {
    __shared__ float sred[256];
    const int i = blockIdx.x;
    const int t = threadIdx.x;
    const float* row = Dm + (size_t)i * NP;

    float mn = 3.4e38f;
    for (int j = t; j < NP; j += 256) mn = fminf(mn, row[j]);
    sred[t] = mn;
    __syncthreads();
    for (int s = 128; s > 0; s >>= 1) {
        if (t < s) sred[t] = fminf(sred[t], sred[t + s]);
        __syncthreads();
    }
    const float dmin = sred[0];
    __syncthreads();
    const float inv = 1.0f / (dmin + EPSF);

    float sumw = 0.0f, maxw = 0.0f;
    for (int j = t; j < NP; j += 256) {
        float w = __expf((1.0f - row[j] * inv) * 2.0f);  // /H_BW, H_BW = 0.5
        sumw += w;
        maxw = fmaxf(maxw, w);
    }
    sred[t] = sumw;
    __syncthreads();
    for (int s = 128; s > 0; s >>= 1) {
        if (t < s) sred[t] += sred[t + s];
        __syncthreads();
    }
    const float sumAll = sred[0];
    __syncthreads();
    sred[t] = maxw;
    __syncthreads();
    for (int s = 128; s > 0; s >>= 1) {
        if (t < s) sred[t] = fmaxf(sred[t], sred[t + s]);
        __syncthreads();
    }
    if (t == 0) {
        float CX = sred[0] / sumAll;
        atomicAdd(loss, -__logf(CX) * (1.0f / (float)NP));
    }
}

// ---------------------------------------------------------------------------
// host-side orchestration
// ---------------------------------------------------------------------------
extern "C" void kernel_launch(void* const* d_in, const int* in_sizes, int n_in,
                              void* d_out, int out_size, void* d_ws, size_t ws_size,
                              hipStream_t stream) {
    (void)in_sizes; (void)n_in; (void)out_size; (void)ws_size;
    const float* target = (const float*)d_in[0];   // (1,128,256,256)
    const float* refer  = (const float*)d_in[1];   // (1,128,256,256)
    const float* mask   = (const float*)d_in[2];   // (1,1,256,256)
    const float* tfield = (const float*)d_in[3];   // (1,64,64,2)
    const float* rfield = (const float*)d_in[4];   // (1,64,64,2)
    float* out = (float*)d_out;

    char* w = (char*)d_ws;
    size_t off = 0;
    auto take = [&](size_t bytes) -> char* {
        char* p = w + off;
        off = (off + bytes + 255) & ~(size_t)255;
        return p;
    };
    int*    tIdx  = (int*)take(NP * sizeof(int));
    int*    rIdx  = (int*)take(NP * sizeof(int));
    float*  ymean = (float*)take(CK * sizeof(float));
    __bf16* Xn    = (__bf16*)take((size_t)NP * CK * sizeof(__bf16));
    __bf16* Yn    = (__bf16*)take((size_t)NP * CK * sizeof(__bf16));
    __bf16* Tm    = (__bf16*)take((size_t)NP * KM * sizeof(__bf16));
    __bf16* Rm    = (__bf16*)take((size_t)NP * KM * sizeof(__bf16));
    float*  xs    = (float*)take(NP * sizeof(float));
    float*  ys    = (float*)take(NP * sizeof(float));
    float*  Dm    = (float*)take((size_t)NP * NP * sizeof(float));

    k_zero<<<1, 32, 0, stream>>>(out);
    k_field_idx<<<NP / 256, 256, 0, stream>>>(tfield, rfield, tIdx, rIdx);
    k_refer_mean<<<(CK + 255) / 256, 256, 0, stream>>>(refer, rIdx, ymean);
    k_gather_norm<<<NP, 256, 0, stream>>>(refer,  rIdx, ymean, Yn);
    k_gather_norm<<<NP, 256, 0, stream>>>(target, tIdx, ymean, Xn);
    k_mask_patches<<<NP / 256, 256, 0, stream>>>(mask, tIdx, rIdx, Tm, Rm, xs, ys);

    dim3 gGemm(NP / BN, NP / BM);   // (N tiles, M tiles)
    k_dist_gemm<<<gGemm, 256, 0, stream>>>(Xn, Yn, Tm, Rm, xs, ys, Dm);

    k_row_reduce<<<NP, 256, 0, stream>>>(Dm, out);
}
